// SignatureRDEBSDE_22832046146333
// MI455X (gfx1250) — compile-verified, matrix operand order checked by reading, not run on hardware
//
#include <hip/hip_runtime.h>
#include <cstdint>
#include <cstddef>

// ---------------------------------------------------------------------------
// Types for CDNA5 WMMA (wave32)
// ---------------------------------------------------------------------------
typedef __bf16 bf16;
typedef bf16  v16bf __attribute__((ext_vector_type(16)));
typedef float v8f   __attribute__((ext_vector_type(8)));

#define DEVINL __device__ __forceinline__

// Problem constants
constexpr int  BB    = 128;
constexpr int  NN    = 64;
constexpr int  DD    = 20;
constexpr int  HH    = 256;
constexpr int  HM    = 1024;          // WM*H
constexpr int  SIGC  = 231;
constexpr int  CC    = 232;
constexpr int  CPAD  = 240;           // padded control channels (15 * 16)
constexpr int  NSTEP = 63;
constexpr float HSTP = 1.0f / 63.0f;
constexpr float EPSL = 1e-5f;

// ---------------------------------------------------------------------------
// Fragment-major layouts (per CDNA5 ISA §7.12.2, wave32):
//   A 16x32 bf16 tile: lane L holds row M=L%16; VGPR v packs K=kbase(v,L/16)+{0,1}
//   B 32x16 bf16 tile: lane L holds col N=L%16; same K packing
// A fragment-major buffer: [m-tile][k-tile][lane(32)][idx(16)]  (32B per lane slot)
// B fragment-major buffer: [n-tile][k-tile][lane(32)][idx(16)]
// => each operand is ONE contiguous 32-byte load per lane (2x global_load_b128).
// kbase(v,half) = 16*(v>>2) + 2*(v&3) + 8*half
// ---------------------------------------------------------------------------
DEVINL int kbase(int v, int half) { return 16 * (v >> 2) + 2 * (v & 3) + 8 * half; }

// inverse map: (row m in tile, k in [0,K)) -> flat fragment-major index
DEVINL size_t afrag_index(int b, int k, int K) {
  int bt = b >> 4, m = b & 15;
  int kt = k >> 5, kk = k & 31;
  int v    = ((kk >> 4) & 1) * 4 + ((kk >> 1) & 3);
  int half = (kk >> 3) & 1;
  int idx  = 2 * v + (kk & 1);
  int lane = half * 16 + m;
  return ((((size_t)bt * (K >> 5)) + kt) << 9) + (lane << 4) + idx;
}

// ---------------------------------------------------------------------------
// One-time weight swizzle: W[K][Nw] row-major f32 -> fragment-major bf16
// ---------------------------------------------------------------------------
__global__ void cvt_w_frag_kernel(const float* __restrict__ W, bf16* __restrict__ Wf,
                                  int K, int Nw) {
  int total = K * Nw;
  int kt_n = K >> 5;
  for (int i = blockIdx.x * blockDim.x + threadIdx.x; i < total; i += gridDim.x * blockDim.x) {
    int idx = i & 15, lane = (i >> 4) & 31, tile = i >> 9;
    int kt = tile % kt_n, nt = tile / kt_n;
    int v = idx >> 1, half = lane >> 4;
    int k = kt * 32 + kbase(v, half) + (idx & 1);
    int n = nt * 16 + (lane & 15);
    Wf[i] = (bf16)W[(size_t)k * Nw + n];
  }
}

// fW4[K=256][H*CC] -> fragment-major padded: [h(256)][cc(15)][kt(8)][lane][16]
__global__ void cvt_w4_frag_kernel(const float* __restrict__ fW4, bf16* __restrict__ W4f) {
  const int total = HH * 15 * 8 * 512;
  for (int i = blockIdx.x * blockDim.x + threadIdx.x; i < total; i += gridDim.x * blockDim.x) {
    int idx = i & 15, lane = (i >> 4) & 31, t = i >> 9;
    int kt = t & 7, cc = (t >> 3) % 15, h = t / (15 * 8);
    int v = idx >> 1, half = lane >> 4;
    int k = kt * 32 + kbase(v, half) + (idx & 1);
    int c = cc * 16 + (lane & 15);
    W4f[i] = (c < CC) ? (bf16)fW4[(size_t)k * (HH * CC) + h * CC + c] : (bf16)0.0f;
  }
}

__global__ void pad_fb4_kernel(const float* __restrict__ fb4, float* __restrict__ fb4p) {
  const int n = HH * CPAD;
  for (int i = blockIdx.x * blockDim.x + threadIdx.x; i < n; i += gridDim.x * blockDim.x) {
    int hh = i / CPAD, c = i % CPAD;
    fb4p[i] = (c < CC) ? fb4[hh * CC + c] : 0.0f;
  }
}

// ---------------------------------------------------------------------------
// bf16 WMMA GEMM, fragment-major operands, no LDS, 3-stage software pipeline:
//   Craw[M=128, Nw] = A[128,K] @ W[K,Nw]   (f32 row-major out)
// grid.x = Nw/16 column tiles; block 256 = 8 waves; wave w owns rows 16w..16w+15.
// Prologue prefetches the whole B panel (prefetch has no LOADcnt side effects),
// steady-state loop is branch-free: load kt+2, compute kt -> WMMA waits keep
// 8 loads in flight (s_wait_loadcnt 0x8), hiding two iterations of L2 latency.
// Requires ktn >= 2 (here ktn is 8 or 32).
// ---------------------------------------------------------------------------
__global__ void gemm_bf16_kernel(const bf16* __restrict__ A, const bf16* __restrict__ Wf,
                                 float* __restrict__ Craw, int K, int Nw) {
  const int n0   = blockIdx.x * 16;
  const int wave = threadIdx.x >> 5;
  const int lane = threadIdx.x & 31;
  const int ktn  = K >> 5;
  const v16bf* Av = (const v16bf*)A  + ((size_t)wave * ktn) * 32 + lane;
  const v16bf* Bv = (const v16bf*)Wf + ((size_t)(n0 >> 4) * ktn) * 32 + lane;

  for (int kt = 0; kt < ktn; ++kt)
    __builtin_prefetch(Bv + (size_t)kt * 32, 0, 1);

  v8f acc = {};
  v16bf a0 = Av[0], b0 = Bv[0];
  v16bf a1 = Av[32], b1 = Bv[32];
  for (int kt = 0; kt < ktn - 2; ++kt) {
    v16bf a2 = Av[(size_t)(kt + 2) * 32];
    v16bf b2 = Bv[(size_t)(kt + 2) * 32];
    acc = __builtin_amdgcn_wmma_f32_16x16x32_bf16(false, a0, false, b0,
                                                  (short)0, acc, false, false);
    a0 = a1; b0 = b1; a1 = a2; b1 = b2;
  }
  acc = __builtin_amdgcn_wmma_f32_16x16x32_bf16(false, a0, false, b0,
                                                (short)0, acc, false, false);
  acc = __builtin_amdgcn_wmma_f32_16x16x32_bf16(false, a1, false, b1,
                                                (short)0, acc, false, false);

  const int half = lane >> 4, n = lane & 15;
  const int row0 = wave * 16;
#pragma unroll
  for (int r = 0; r < 8; ++r)
    Craw[(size_t)(row0 + r + 8 * half) * Nw + n0 + n] = acc[r];
}

// ---------------------------------------------------------------------------
// Fused last layer: gout[b,h] = sum_c ( (a3 @ fW4)[b,h,c] + fb4[h,c] ) * dX[b,c]
// A3f fragment-major [bt][kt(8)][lane][16]; W4f [h][cc(15)][kt(8)][lane][16].
// grid = (H, B/16); block 256 (8 waves). A fragments hoisted to registers;
// B stream software-pipelined; wave w covers c-chunks {w, w+8}; epilogue
// multiplies by dX and row-reduces (shuffle + LDS cross-wave).
// ---------------------------------------------------------------------------
__global__ void vf4_contract_kernel(const bf16* __restrict__ A3f,
                                    const bf16* __restrict__ W4f,
                                    const float* __restrict__ fb4p /*[H][CPAD]*/,
                                    const float* __restrict__ dX /*[B][CC]*/,
                                    float* __restrict__ gout /*[B][H]*/) {
  const int h    = blockIdx.x;
  const int b0   = blockIdx.y * 16;
  const int wave = threadIdx.x >> 5;
  const int lane = threadIdx.x & 31;
  const int half = lane >> 4, nlo = lane & 15;

  __shared__ float dXs[16 * CPAD];     // zero-padded dX tile
  __shared__ float psum[8][16];

  for (int i = threadIdx.x; i < 16 * CPAD; i += 256) {
    int bb = i / CPAD, c = i % CPAD;
    dXs[i] = (c < CC) ? dX[(size_t)(b0 + bb) * CC + c] : 0.0f;
  }
  __syncthreads();

  // hoist the 8 A fragments for this b-tile (shared across all 15 c-chunks)
  const v16bf* Av = (const v16bf*)A3f + ((size_t)(b0 >> 4) * 8) * 32 + lane;
  v16bf areg[8];
#pragma unroll
  for (int kt = 0; kt < 8; ++kt) areg[kt] = Av[(size_t)kt * 32];

  float rsum[8];
#pragma unroll
  for (int r = 0; r < 8; ++r) rsum[r] = 0.0f;

  for (int cc = wave; cc < CPAD / 16; cc += 8) {
    const v16bf* Bv = (const v16bf*)W4f + (((size_t)h * 15 + cc) * 8) * 32 + lane;
#pragma unroll
    for (int kt = 0; kt < 8; ++kt)
      __builtin_prefetch(Bv + (size_t)kt * 32, 0, 1);
    v8f acc = {};
    v16bf b_cur = Bv[0];
#pragma unroll
    for (int kt = 0; kt < 7; ++kt) {
      v16bf b_nxt = Bv[(size_t)(kt + 1) * 32];
      acc = __builtin_amdgcn_wmma_f32_16x16x32_bf16(false, areg[kt], false, b_cur,
                                                    (short)0, acc, false, false);
      b_cur = b_nxt;
    }
    acc = __builtin_amdgcn_wmma_f32_16x16x32_bf16(false, areg[7], false, b_cur,
                                                  (short)0, acc, false, false);
    float bias = fb4p[h * CPAD + cc * 16 + nlo];
#pragma unroll
    for (int r = 0; r < 8; ++r) {
      float dx = dXs[(r + 8 * half) * CPAD + cc * 16 + nlo];
      rsum[r] += (acc[r] + bias) * dx;
    }
  }
  // reduce over the 16 column lanes of each half
#pragma unroll
  for (int r = 0; r < 8; ++r) {
#pragma unroll
    for (int m = 8; m >= 1; m >>= 1) rsum[r] += __shfl_xor(rsum[r], m, 32);
  }
  if ((lane & 15) == 0) {
#pragma unroll
    for (int r = 0; r < 8; ++r) psum[wave][r + 8 * half] = rsum[r];
  }
  __syncthreads();
  if (threadIdx.x < 16) {
    float s = 0.0f;
#pragma unroll
    for (int w = 0; w < 8; ++w) s += psum[w][threadIdx.x];
    gout[(size_t)(b0 + threadIdx.x) * HH + h] = s;
  }
}

// ---------------------------------------------------------------------------
// LayerNorm(raw + bias)*gamma + beta -> relu -> bf16 written fragment-major.
// ---------------------------------------------------------------------------
template <int NCH>
__global__ void ln_relu_bf16_kernel(const float* __restrict__ raw, const float* __restrict__ bias,
                                    const float* __restrict__ gam, const float* __restrict__ bet,
                                    bf16* __restrict__ outbf) {
  constexpr int PER = NCH / 256;
  const int b = blockIdx.x;
  const float* row = raw + (size_t)b * NCH;
  float v[PER], ls = 0.0f, lq = 0.0f;
#pragma unroll
  for (int i = 0; i < PER; ++i) {
    int c = threadIdx.x + 256 * i;
    v[i] = row[c] + bias[c];
    ls += v[i]; lq += v[i] * v[i];
  }
  __shared__ float s1[256], s2[256];
  s1[threadIdx.x] = ls; s2[threadIdx.x] = lq;
  __syncthreads();
  for (int st = 128; st > 0; st >>= 1) {
    if (threadIdx.x < st) { s1[threadIdx.x] += s1[threadIdx.x + st]; s2[threadIdx.x] += s2[threadIdx.x + st]; }
    __syncthreads();
  }
  float mean = s1[0] / NCH;
  float var  = s2[0] / NCH - mean * mean;
  float rstd = rsqrtf(var + EPSL);
#pragma unroll
  for (int i = 0; i < PER; ++i) {
    int c = threadIdx.x + 256 * i;
    outbf[afrag_index(b, c, NCH)] = (bf16)fmaxf((v[i] - mean) * rstd * gam[c] + bet[c], 0.0f);
  }
}

__global__ void bias_relu_bf16_kernel(const float* __restrict__ raw, const float* __restrict__ bias,
                                      bf16* __restrict__ outbf, int ncols, int n) {
  for (int i = blockIdx.x * blockDim.x + threadIdx.x; i < n; i += gridDim.x * blockDim.x) {
    int b = i / ncols, c = i % ncols;
    outbf[afrag_index(b, c, ncols)] = (bf16)fmaxf(raw[i] + bias[c], 0.0f);
  }
}

__global__ void axpy_bf_kernel(const float* __restrict__ z, const float* __restrict__ g,
                               float s, float* __restrict__ zo, bf16* __restrict__ zob, int n) {
  for (int i = blockIdx.x * blockDim.x + threadIdx.x; i < n; i += gridDim.x * blockDim.x) {
    float v = z[i] + s * g[i];
    zo[i] = v;
    zob[afrag_index(i >> 8, i & 255, HH)] = (bf16)v;
  }
}

// ---------------------------------------------------------------------------
// Depth-2 streamed log-signature: 1 thread per (b, channel)
// ---------------------------------------------------------------------------
__global__ void logsig_kernel(const float* __restrict__ x, float* __restrict__ out) {
  int b = blockIdx.x, s = threadIdx.x;
  if (s >= SIGC) return;
  if (s < 21) {
    for (int n = 0; n < NN; ++n) {
      float v = (s == 0) ? n * HSTP : x[((size_t)b * NN + n) * DD + (s - 1)];
      out[((size_t)b * NN + n) * SIGC + s] = v;
    }
  } else {
    int r = s - 21, i = 0;
    while (r >= 20 - i) { r -= 20 - i; ++i; }
    int j = i + 1 + r;
    float pip = 0.0f, pjp = 0.0f, acc = 0.0f;
    for (int n = 0; n < NN; ++n) {
      float pi = (i == 0) ? n * HSTP : x[((size_t)b * NN + n) * DD + (i - 1)];
      float pj = x[((size_t)b * NN + n) * DD + (j - 1)];  // j >= 1 always
      float di = pi - pip, dj = pj - pjp;
      acc += 0.5f * (pip * dj - di * pjp);
      out[((size_t)b * NN + n) * SIGC + s] = acc;
      pip = pi; pjp = pj;
    }
  }
}

// sig_norm + time augmentation: one block per (b,n)
__global__ void signorm_kernel(const float* __restrict__ ls, const float* __restrict__ sg,
                               const float* __restrict__ sb, float* __restrict__ y) {
  int bn = blockIdx.x;
  const float* row = ls + (size_t)bn * SIGC;
  float v = (threadIdx.x < SIGC) ? row[threadIdx.x] : 0.0f;
  __shared__ float s1[256], s2[256];
  s1[threadIdx.x] = v; s2[threadIdx.x] = v * v;
  __syncthreads();
  for (int st = 128; st > 0; st >>= 1) {
    if (threadIdx.x < st) { s1[threadIdx.x] += s1[threadIdx.x + st]; s2[threadIdx.x] += s2[threadIdx.x + st]; }
    __syncthreads();
  }
  float mean = s1[0] / SIGC;
  float var  = s2[0] / SIGC - mean * mean;
  float rstd = rsqrtf(var + EPSL);
  if (threadIdx.x < SIGC)
    y[(size_t)bn * CC + 1 + threadIdx.x] = (v - mean) * rstd * sg[threadIdx.x] + sb[threadIdx.x];
  if (threadIdx.x == 0)
    y[(size_t)bn * CC] = (bn % NN) * HSTP;
}

// ---------------------------------------------------------------------------
// Natural cubic spline (Thomas solve per (b,c)) + knot/midpoint derivatives.
// Output layout [n][b][c] so each scan step reads a contiguous dX slab.
// ---------------------------------------------------------------------------
__global__ void spline_kernel(const float* __restrict__ y, float* __restrict__ dknot,
                              float* __restrict__ dmid) {
  int idx = blockIdx.x * blockDim.x + threadIdx.x;
  if (idx >= BB * CC) return;
  int b = idx / CC, c = idx % CC;
  const float h = HSTP, a = h / 6.0f, bd = 2.0f * h / 3.0f;
  float yv[NN], dp[NN], cp[NN], M[NN];
  for (int n = 0; n < NN; ++n) yv[n] = y[((size_t)b * NN + n) * CC + c];
  cp[0] = 0.0f; dp[0] = 0.0f;
  for (int n = 1; n <= NN - 2; ++n) {
    float rhs = (yv[n + 1] - 2.0f * yv[n] + yv[n - 1]) / h;
    float m = bd - a * cp[n - 1];
    cp[n] = a / m;
    dp[n] = (rhs - a * dp[n - 1]) / m;
  }
  M[NN - 1] = 0.0f;
  for (int n = NN - 2; n >= 0; --n) M[n] = dp[n] - cp[n] * M[n + 1];
  for (int n = 0; n < NN - 1; ++n) {
    float dy = (yv[n + 1] - yv[n]) / h;
    size_t o = ((size_t)n * BB + b) * CC + c;
    dknot[o] = dy - (h / 6.0f) * (2.0f * M[n] + M[n + 1]);
    dmid[o]  = dy + (h / 24.0f) * (M[n] - M[n + 1]);
  }
}

// ---------------------------------------------------------------------------
// Initial hidden state + output heads (tiny GEMMs, plain VALU)
// ---------------------------------------------------------------------------
__global__ void h0_kernel(const float* __restrict__ x, const float* __restrict__ hW1,
                          const float* __restrict__ hb1, const float* __restrict__ hW2,
                          const float* __restrict__ hb2, float* __restrict__ z,
                          bf16* __restrict__ zbf) {
  int b = blockIdx.x, j = threadIdx.x;
  __shared__ float a[HH];
  float s = hb1[j];
  for (int k = 0; k < DD; ++k) s += x[(size_t)b * NN * DD + k] * hW1[k * HH + j];
  a[j] = fmaxf(s, 0.0f);
  __syncthreads();
  float o = hb2[j];
  for (int k = 0; k < HH; ++k) o += a[k] * hW2[k * HH + j];
  z[(size_t)b * HH + j] = o;
  zbf[afrag_index(b, j, HH)] = (bf16)o;
}

__global__ void headY_kernel(const float* __restrict__ hT, const float* __restrict__ yW1,
                             const float* __restrict__ yb1, const float* __restrict__ yW2,
                             const float* __restrict__ yb2, float* __restrict__ out) {
  int b = blockIdx.x, j = threadIdx.x;  // 128 threads
  __shared__ float red[128];
  float s = yb1[j];
  for (int k = 0; k < HH; ++k) s += hT[(size_t)b * HH + k] * yW1[k * 128 + j];
  red[j] = fmaxf(s, 0.0f) * yW2[j];
  __syncthreads();
  for (int st = 64; st > 0; st >>= 1) {
    if (j < st) red[j] += red[j + st];
    __syncthreads();
  }
  if (j == 0) out[b] = red[0] + yb2[0];
}

__global__ void headZ_kernel(const float* __restrict__ hT, const float* __restrict__ zW1,
                             const float* __restrict__ zb1, const float* __restrict__ zW2,
                             const float* __restrict__ zb2, float* __restrict__ out) {
  int b = blockIdx.x, j = threadIdx.x;  // 256 threads
  __shared__ float u[HH];
  float s = zb1[j];
  for (int k = 0; k < HH; ++k) s += hT[(size_t)b * HH + k] * zW1[k * HH + j];
  u[j] = fmaxf(s, 0.0f);
  __syncthreads();
  if (j < DD) {
    float o = zb2[j];
    for (int k = 0; k < HH; ++k) o += u[k] * zW2[k * DD + j];
    out[(size_t)b * DD + j] = o;
  }
}

// ---------------------------------------------------------------------------
// Host orchestration
// ---------------------------------------------------------------------------
extern "C" void kernel_launch(void* const* d_in, const int* in_sizes, int n_in,
                              void* d_out, int out_size, void* d_ws, size_t ws_size,
                              hipStream_t stream) {
  (void)in_sizes; (void)n_in; (void)out_size; (void)ws_size;
  const float* x   = (const float*)d_in[0];
  const float* hW1 = (const float*)d_in[1];
  const float* hb1 = (const float*)d_in[2];
  const float* hW2 = (const float*)d_in[3];
  const float* hb2 = (const float*)d_in[4];
  const float* fW1 = (const float*)d_in[5];
  const float* fb1 = (const float*)d_in[6];
  const float* g1  = (const float*)d_in[7];
  const float* be1 = (const float*)d_in[8];
  const float* fW2 = (const float*)d_in[9];
  const float* fb2 = (const float*)d_in[10];
  const float* g2  = (const float*)d_in[11];
  const float* be2 = (const float*)d_in[12];
  const float* fW3 = (const float*)d_in[13];
  const float* fb3 = (const float*)d_in[14];
  const float* fW4 = (const float*)d_in[15];
  const float* fb4 = (const float*)d_in[16];
  const float* yW1 = (const float*)d_in[17];
  const float* yb1 = (const float*)d_in[18];
  const float* yW2 = (const float*)d_in[19];
  const float* yb2 = (const float*)d_in[20];
  const float* zW1 = (const float*)d_in[21];
  const float* zb1 = (const float*)d_in[22];
  const float* zW2 = (const float*)d_in[23];
  const float* zb2 = (const float*)d_in[24];
  const float* sg  = (const float*)d_in[25];
  const float* sb  = (const float*)d_in[26];
  float* dout = (float*)d_out;

  // workspace carve (256-byte aligned segments)
  size_t off = 0;
  auto carve = [&](size_t bytes) -> void* {
    void* p = (char*)d_ws + off;
    off += (bytes + 255) & ~(size_t)255;
    return p;
  };
  bf16*  W1f    = (bf16*)carve((size_t)HH * HM * 2);
  bf16*  W2f    = (bf16*)carve((size_t)HM * HH * 2);
  bf16*  W3f    = (bf16*)carve((size_t)HH * HH * 2);
  bf16*  W4f    = (bf16*)carve((size_t)HH * 15 * 8 * 512 * 2);
  float* fb4p   = (float*)carve((size_t)HH * CPAD * 4);
  float* lsbuf  = (float*)carve((size_t)BB * NN * SIGC * 4);
  float* ybuf   = (float*)carve((size_t)BB * NN * CC * 4);
  float* dknot  = (float*)carve((size_t)NSTEP * BB * CC * 4);
  float* dmid   = (float*)carve((size_t)NSTEP * BB * CC * 4);
  float* zbuf   = (float*)carve((size_t)BB * HH * 4);
  bf16*  zbfb   = (bf16*)carve((size_t)BB * HH * 2);
  float* ztmp   = (float*)carve((size_t)BB * HH * 4);
  bf16*  ztmpbf = (bf16*)carve((size_t)BB * HH * 2);
  float* a1raw  = (float*)carve((size_t)BB * HM * 4);
  bf16*  a1bf   = (bf16*)carve((size_t)BB * HM * 2);
  float* a2raw  = (float*)carve((size_t)BB * HH * 4);
  bf16*  a2bf   = (bf16*)carve((size_t)BB * HH * 2);
  bf16*  a3bf   = (bf16*)carve((size_t)BB * HH * 2);
  float* gbuf1  = (float*)carve((size_t)BB * HH * 4);
  float* gbuf2  = (float*)carve((size_t)BB * HH * 4);

  // --- one-time weight swizzle to fragment-major bf16 ---
  cvt_w_frag_kernel<<<512, 256, 0, stream>>>(fW1, W1f, HH, HM);
  cvt_w_frag_kernel<<<512, 256, 0, stream>>>(fW2, W2f, HM, HH);
  cvt_w_frag_kernel<<<128, 256, 0, stream>>>(fW3, W3f, HH, HH);
  cvt_w4_frag_kernel<<<4096, 256, 0, stream>>>(fW4, W4f);
  pad_fb4_kernel<<<240, 256, 0, stream>>>(fb4, fb4p);

  // --- log-signature -> sig_norm -> spline derivatives ---
  logsig_kernel<<<BB, 256, 0, stream>>>(x, lsbuf);
  signorm_kernel<<<BB * NN, 256, 0, stream>>>(lsbuf, sg, sb, ybuf);
  spline_kernel<<<(BB * CC + 255) / 256, 256, 0, stream>>>(ybuf, dknot, dmid);

  // --- initial hidden state ---
  h0_kernel<<<BB, HH, 0, stream>>>(x, hW1, hb1, hW2, hb2, zbuf, zbfb);

  // --- one vector-field evaluation g(z, dX) ---
  auto geval = [&](const bf16* zin_bf, const float* dX, float* gout) {
    gemm_bf16_kernel<<<HM / 16, 256, 0, stream>>>(zin_bf, W1f, a1raw, HH, HM);
    ln_relu_bf16_kernel<HM><<<BB, 256, 0, stream>>>(a1raw, fb1, g1, be1, a1bf);
    gemm_bf16_kernel<<<HH / 16, 256, 0, stream>>>(a1bf, W2f, a2raw, HM, HH);
    ln_relu_bf16_kernel<HH><<<BB, 256, 0, stream>>>(a2raw, fb2, g2, be2, a2bf);
    gemm_bf16_kernel<<<HH / 16, 256, 0, stream>>>(a2bf, W3f, a2raw, HH, HH);
    bias_relu_bf16_kernel<<<128, 256, 0, stream>>>(a2raw, fb3, a3bf, HH, BB * HH);
    vf4_contract_kernel<<<dim3(HH, BB / 16), 256, 0, stream>>>(a3bf, W4f, fb4p, dX, gout);
  };

  // --- explicit midpoint scan ---
  for (int n = 0; n < NSTEP; ++n) {
    const float* dk = dknot + (size_t)n * BB * CC;
    const float* dm = dmid  + (size_t)n * BB * CC;
    geval(zbfb, dk, gbuf1);
    axpy_bf_kernel<<<128, 256, 0, stream>>>(zbuf, gbuf1, 0.5f * HSTP, ztmp, ztmpbf, BB * HH);
    geval(ztmpbf, dm, gbuf2);
    axpy_bf_kernel<<<128, 256, 0, stream>>>(zbuf, gbuf2, HSTP, zbuf, zbfb, BB * HH);
  }

  // --- heads + terminal state ---
  headY_kernel<<<BB, 128, 0, stream>>>(zbuf, yW1, yb1, yW2, yb2, dout);
  headZ_kernel<<<BB, 256, 0, stream>>>(zbuf, zW1, zb1, zW2, zb2, dout + BB);
  hipMemcpyAsync(dout + BB + BB * DD, zbuf, (size_t)BB * HH * sizeof(float),
                 hipMemcpyDeviceToDevice, stream);
}
